// AttnDecoderCell_34308198760514
// MI455X (gfx1250) — matrix operand order repaired and checked by reading, not compiled, not run on hardware
//
#include <hip/hip_runtime.h>
#include <cstdint>
#include <cstddef>

// ---------------------------------------------------------------------------
// AttnDecoderCell for MI455X (gfx1250, wave32, WMMA).
// B=512, L=256, D=DIN=1024.
// HBM-bound on constants (512MB -> ~22us at 23.3 TB/s); attention kernel
// streams it once from HBM (second pass hits the per-batch 1MB slice in L2).
// GEMMs run as bf16 WMMA 16x16x32 (f32 accum). Weights are converted once
// fp32 -> bf16 *pre-swizzled into the WMMA B-fragment lane layout*, so the
// GEMM inner loop is 4x global_load_b128 per v_wmma (no scalar u16 loads).
// ---------------------------------------------------------------------------

#define DEV_INLINE __device__ __forceinline__

typedef __attribute__((ext_vector_type(16))) __bf16 v16bf;
typedef __attribute__((ext_vector_type(8)))  float  v8f;

static constexpr int kB   = 512;
static constexpr int kL   = 256;
static constexpr int kD   = 1024;
static constexpr int kDIN = 1024;

struct alignas(8) US4 { unsigned short x, y, z, w; };

DEV_INLINE unsigned short f32_to_bf16(float f) {
  union { float f; uint32_t u; } c; c.f = f;
  uint32_t r = c.u + 0x7FFFu + ((c.u >> 16) & 1u);   // round-to-nearest-even
  return (unsigned short)(r >> 16);
}

DEV_INLINE float sigmoidf_fast(float x) {
  return 1.0f / (1.0f + __expf(-x));
}

// ---------------------------------------------------------------------------
// Weight conversion fp32 [K,N=1024] -> bf16, pre-swizzled to the CDNA5 WMMA
// B-operand lane layout (32x16 bf16 tile, wave32):
//   lane = hi*16 + n ; element e holds W[kt*32 + hi*8 + (e<8 ? e : e+8)][ct*16 + n]
//   swizzled offset = ((kt*(N/16) + ct)*32 + lane)*16 + e   (elems, bf16)
// Each thread produces one lane-chunk (16 elems = 32B, stored as 2x uint4).
// ---------------------------------------------------------------------------
__global__ __launch_bounds__(256) void cvt_swizzle_b_kernel(
    const float* __restrict__ W, unsigned short* __restrict__ sw, int K) {
  const int N = kD;
  const int t = blockIdx.x * 256 + threadIdx.x;   // t = tile*32 + lane
  const int total = (K / 32) * (N / 16) * 32;
  if (t >= total) return;
  const int lane = t & 31;
  const int tile = t >> 5;
  const int ct = tile % (N / 16);
  const int kt = tile / (N / 16);
  const int hi = lane >> 4, n = lane & 15;
  const int k0 = kt * 32 + hi * 8;
  const int col = ct * 16 + n;

  alignas(16) unsigned short buf[16];
#pragma unroll
  for (int e = 0; e < 16; ++e) {
    buf[e] = f32_to_bf16(W[(size_t)(k0 + (e < 8 ? e : e + 8)) * N + col]);
  }
  uint4* dst = (uint4*)(sw + (size_t)t * 16);
  dst[0] = *(const uint4*)&buf[0];
  dst[1] = *(const uint4*)&buf[8];
}

// ---------------------------------------------------------------------------
// Attention + context kernel: one block (256 thr = 8 waves) per batch row.
//  energy[l] = state.w_att[0:D] + constants[b,l,:].w_att[D:2D] + b_att
//  alpha = softmax_L(energy);  v = alpha^T constants[b]
// Also emits bf16 state and bf16 inp=[x, v] for the WMMA GEMMs.
// ---------------------------------------------------------------------------
__global__ __launch_bounds__(256) void attn_ctx_kernel(
    const float* __restrict__ x, const float* __restrict__ state,
    const float* __restrict__ constants, const float* __restrict__ w_att,
    const float* __restrict__ b_att,
    float* __restrict__ v_out,
    unsigned short* __restrict__ state_bf,
    unsigned short* __restrict__ inp_bf) {
  const int b    = blockIdx.x;
  const int tid  = threadIdx.x;
  const int lane = tid & 31;
  const int wave = tid >> 5;

  __shared__ float wc[kD];       // w_att upper half (constants part)
  __shared__ float energy[kL];   // energies -> alpha
  __shared__ float red[8];
  __shared__ float bcast;

  // Stage w_att[D:2D) into LDS.
  for (int i = tid; i < kD; i += 256) wc[i] = w_att[kD + i];

  // e_state = dot(state[b], w_att[0:D]) (softmax-invariant, kept for fidelity)
  float es;
  {
    const float4 s4 = ((const float4*)(state + (size_t)b * kD))[tid];
    const float4 w4 = ((const float4*)w_att)[tid];
    es = s4.x * w4.x + s4.y * w4.y + s4.z * w4.z + s4.w * w4.w;
  }
  for (int m = 16; m > 0; m >>= 1) es += __shfl_xor(es, m, 32);
  if (lane == 0) red[wave] = es;
  __syncthreads();
  if (tid == 0) {
    float t = 0.f;
    for (int i = 0; i < 8; ++i) t += red[i];
    bcast = t + b_att[0];
  }
  __syncthreads();
  const float ebase = bcast;

  // Pass 1: energies. Wave w handles rows l = w, w+8, ..., lanes split D.
  const float* cb = constants + (size_t)b * kL * kD;
  for (int t = 0; t < kL / 8; ++t) {
    const int l = wave + 8 * t;
    const float4* row = (const float4*)(cb + (size_t)l * kD);
    if (t + 1 < kL / 8) {   // keep the HBM stream ahead (global_prefetch_b8)
      __builtin_prefetch(cb + (size_t)(l + 8) * kD + lane * 32, 0, 1);
    }
    float p = 0.f;
#pragma unroll
    for (int i = 0; i < 8; ++i) {
      const float4 c4 = row[lane + 32 * i];
      const float4 w4 = ((const float4*)wc)[lane + 32 * i];
      p += c4.x * w4.x + c4.y * w4.y + c4.z * w4.z + c4.w * w4.w;
    }
    for (int m = 16; m > 0; m >>= 1) p += __shfl_xor(p, m, 32);
    if (lane == 0) energy[l] = p + ebase;
  }
  __syncthreads();

  // Softmax over L=256 (one element per thread).
  float e = energy[tid];
  float mx = e;
  for (int m = 16; m > 0; m >>= 1) mx = fmaxf(mx, __shfl_xor(mx, m, 32));
  if (lane == 0) red[wave] = mx;
  __syncthreads();
  if (tid == 0) {
    float t = red[0];
    for (int i = 1; i < 8; ++i) t = fmaxf(t, red[i]);
    bcast = t;
  }
  __syncthreads();
  const float ex = __expf(e - bcast);
  float sm = ex;
  for (int m = 16; m > 0; m >>= 1) sm += __shfl_xor(sm, m, 32);
  if (lane == 0) red[wave] = sm;
  __syncthreads();
  if (tid == 0) {
    float t = 0.f;
    for (int i = 0; i < 8; ++i) t += red[i];
    bcast = t;
  }
  __syncthreads();
  energy[tid] = ex / bcast;   // alpha
  __syncthreads();

  // Pass 2: v[b, 4t..4t+3] = sum_l alpha[l] * constants[b, l, 4t..].
  // The 1MB slice was just streamed -> L2 hit.
  float4 acc = {0.f, 0.f, 0.f, 0.f};
  const float4* c4b = (const float4*)cb;
  for (int l = 0; l < kL; ++l) {
    const float al = energy[l];
    const float4 c4 = c4b[l * (kD / 4) + tid];
    acc.x += al * c4.x; acc.y += al * c4.y;
    acc.z += al * c4.z; acc.w += al * c4.w;
  }
  ((float4*)(v_out + (size_t)b * kD))[tid] = acc;

  // inp_bf = [bf16(x), bf16(v)], state_bf = bf16(state)
  const float4 xv = ((const float4*)(x + (size_t)b * kDIN))[tid];
  const float4 sv = ((const float4*)(state + (size_t)b * kD))[tid];
  unsigned short* ip = inp_bf + (size_t)b * (kDIN + kD);
  US4 u;
  u.x = f32_to_bf16(xv.x); u.y = f32_to_bf16(xv.y);
  u.z = f32_to_bf16(xv.z); u.w = f32_to_bf16(xv.w);
  *(US4*)(ip + 4 * tid) = u;
  u.x = f32_to_bf16(acc.x); u.y = f32_to_bf16(acc.y);
  u.z = f32_to_bf16(acc.z); u.w = f32_to_bf16(acc.w);
  *(US4*)(ip + kDIN + 4 * tid) = u;
  u.x = f32_to_bf16(sv.x); u.y = f32_to_bf16(sv.y);
  u.z = f32_to_bf16(sv.z); u.w = f32_to_bf16(sv.w);
  *(US4*)(state_bf + (size_t)b * kD + 4 * tid) = u;
}

// ---------------------------------------------------------------------------
// WMMA fragment loads (CDNA5 16-bit operand layouts, wave32).
// A 16x32: lane 0-15 -> M=lane, K elems {0..7,16..23}; lanes 16-31 same rows,
//          K shifted by +8. Per-lane half-rows contiguous -> 2x b128 load.
// B 32x16: read from the pre-swizzled weight image -> 2x b128, coalesced
//          (wave reads 1KB contiguous per tile).
// C/D:     M = v + 8*(lane>=16), N = lane&15.
// ---------------------------------------------------------------------------
union BfTile { v16bf v; uint4 q[2]; };

DEV_INLINE v16bf load_a_tile(const unsigned short* __restrict__ A, int lda,
                             int row, int kb, int hi) {
  const unsigned short* p = A + (size_t)row * lda + kb + hi * 8;
  BfTile t;
  t.q[0] = *(const uint4*)p;          // elems 0..7  : K = kb+8*hi + 0..7
  t.q[1] = *(const uint4*)(p + 16);   // elems 8..15 : K = kb+8*hi + 16..23
  return t.v;
}

DEV_INLINE v16bf load_b_sw(const unsigned short* __restrict__ Wsw,
                           int kt, int ct, int lane) {
  const unsigned short* p =
      Wsw + (((size_t)kt * (kD / 16) + ct) * 32 + lane) * 16;
  BfTile t;
  t.q[0] = *(const uint4*)p;          // elems 0..7
  t.q[1] = *(const uint4*)(p + 8);    // elems 8..15
  return t.v;
}

// ---------------------------------------------------------------------------
// Gates z and r:  pre = [state;inp](B x 3072) @ [w;u](3072 x D) + bias
//   gate==0: z = sigmoid(pre)               -> z_f32
//   gate==1: r = sigmoid(pre); sr = state*r -> sr_bf
// Each wave owns one 16x16 tile; block = 8 waves = 16x128 strip.
// grid = (D/128, B/16, 2)
// ---------------------------------------------------------------------------
__global__ __launch_bounds__(256) void gates_zr_kernel(
    const unsigned short* __restrict__ state_bf,
    const unsigned short* __restrict__ inp_bf,
    const unsigned short* __restrict__ wz_sw, const unsigned short* __restrict__ uz_sw,
    const unsigned short* __restrict__ wr_sw, const unsigned short* __restrict__ ur_sw,
    const float* __restrict__ b_z, const float* __restrict__ b_r,
    const float* __restrict__ state_f32,
    float* __restrict__ z_out, unsigned short* __restrict__ sr_bf) {
  const int lane = threadIdx.x & 31;
  const int wave = threadIdx.x >> 5;
  const int gate = blockIdx.z;                 // uniform
  const int col0 = blockIdx.x * 128 + wave * 16;
  const int row0 = blockIdx.y * 16;
  const int hi = lane >> 4, n = lane & 15;
  const int arow = row0 + n;
  const int col  = col0 + n;
  const int ct   = col0 >> 4;

  const unsigned short* Wsw = gate ? wr_sw : wz_sw;
  const unsigned short* Usw = gate ? ur_sw : uz_sw;
  const float* bias = gate ? b_r : b_z;

  v8f acc = {};
  for (int kt = 0; kt < kD / 32; ++kt) {             // state @ w
    v16bf a  = load_a_tile(state_bf, kD, arow, kt * 32, hi);
    v16bf bb = load_b_sw(Wsw, kt, ct, lane);
    acc = __builtin_amdgcn_wmma_f32_16x16x32_bf16(false, a, false, bb,
                                                  (short)0, acc, false, false);
  }
  for (int kt = 0; kt < (kDIN + kD) / 32; ++kt) {    // inp @ u
    v16bf a  = load_a_tile(inp_bf, kDIN + kD, arow, kt * 32, hi);
    v16bf bb = load_b_sw(Usw, kt, ct, lane);
    acc = __builtin_amdgcn_wmma_f32_16x16x32_bf16(false, a, false, bb,
                                                  (short)0, acc, false, false);
  }

  const float bv = bias[col];
#pragma unroll
  for (int v = 0; v < 8; ++v) {
    const int row = row0 + v + 8 * hi;
    const size_t idx = (size_t)row * kD + col;
    const float g = sigmoidf_fast(acc[v] + bv);
    if (gate == 0) {
      z_out[idx] = g;
    } else {
      sr_bf[idx] = f32_to_bf16(state_f32[idx] * g);
    }
  }
}

// ---------------------------------------------------------------------------
// h + output:  pre = [sr;inp] @ [w_h;u_h] + b_h ; h = tanh(pre)
//              out = (1-z)*h + z*state + v
// grid = (D/128, B/16)
// ---------------------------------------------------------------------------
__global__ __launch_bounds__(256) void final_h_kernel(
    const unsigned short* __restrict__ sr_bf,
    const unsigned short* __restrict__ inp_bf,
    const unsigned short* __restrict__ wh_sw, const unsigned short* __restrict__ uh_sw,
    const float* __restrict__ b_h,
    const float* __restrict__ z_f32, const float* __restrict__ state_f32,
    const float* __restrict__ v_f32, float* __restrict__ out) {
  const int lane = threadIdx.x & 31;
  const int wave = threadIdx.x >> 5;
  const int col0 = blockIdx.x * 128 + wave * 16;
  const int row0 = blockIdx.y * 16;
  const int hi = lane >> 4, n = lane & 15;
  const int arow = row0 + n;
  const int col  = col0 + n;
  const int ct   = col0 >> 4;

  v8f acc = {};
  for (int kt = 0; kt < kD / 32; ++kt) {             // (state*r) @ w_h
    v16bf a  = load_a_tile(sr_bf, kD, arow, kt * 32, hi);
    v16bf bb = load_b_sw(wh_sw, kt, ct, lane);
    acc = __builtin_amdgcn_wmma_f32_16x16x32_bf16(false, a, false, bb,
                                                  (short)0, acc, false, false);
  }
  for (int kt = 0; kt < (kDIN + kD) / 32; ++kt) {    // inp @ u_h
    v16bf a  = load_a_tile(inp_bf, kDIN + kD, arow, kt * 32, hi);
    v16bf bb = load_b_sw(uh_sw, kt, ct, lane);
    acc = __builtin_amdgcn_wmma_f32_16x16x32_bf16(false, a, false, bb,
                                                  (short)0, acc, false, false);
  }

  const float bv = b_h[col];
#pragma unroll
  for (int v = 0; v < 8; ++v) {
    const int row = row0 + v + 8 * hi;
    const size_t idx = (size_t)row * kD + col;
    const float h = tanhf(acc[v] + bv);
    const float z = z_f32[idx];
    out[idx] = (1.0f - z) * h + z * state_f32[idx] + v_f32[idx];
  }
}

// ---------------------------------------------------------------------------
// Launch
// ---------------------------------------------------------------------------
extern "C" void kernel_launch(void* const* d_in, const int* in_sizes, int n_in,
                              void* d_out, int out_size, void* d_ws, size_t ws_size,
                              hipStream_t stream) {
  const float* x         = (const float*)d_in[0];
  const float* state     = (const float*)d_in[1];
  const float* constants = (const float*)d_in[2];
  const float* w_att     = (const float*)d_in[3];
  const float* b_att     = (const float*)d_in[4];
  const float* w_z       = (const float*)d_in[5];
  const float* u_z       = (const float*)d_in[6];
  const float* b_z       = (const float*)d_in[7];
  const float* w_r       = (const float*)d_in[8];
  const float* u_r       = (const float*)d_in[9];
  const float* b_r       = (const float*)d_in[10];
  const float* w_h       = (const float*)d_in[11];
  const float* u_h       = (const float*)d_in[12];
  const float* b_h       = (const float*)d_in[13];
  float* out = (float*)d_out;

  // Workspace layout (bytes), total ~26 MB.
  char* ws = (char*)d_ws;
  float*          v_f32    = (float*)(ws + 0);                     // 2 MB
  float*          z_f32    = (float*)(ws + (2u << 20));            // 2 MB
  unsigned short* state_bf = (unsigned short*)(ws + (4u << 20));   // 1 MB
  unsigned short* sr_bf    = (unsigned short*)(ws + (5u << 20));   // 1 MB
  unsigned short* inp_bf   = (unsigned short*)(ws + (6u << 20));   // 2 MB
  unsigned short* wz_sw    = (unsigned short*)(ws + (8u << 20));   // 2 MB
  unsigned short* wr_sw    = (unsigned short*)(ws + (10u << 20));  // 2 MB
  unsigned short* wh_sw    = (unsigned short*)(ws + (12u << 20));  // 2 MB
  unsigned short* uz_sw    = (unsigned short*)(ws + (14u << 20));  // 4 MB
  unsigned short* ur_sw    = (unsigned short*)(ws + (18u << 20));  // 4 MB
  unsigned short* uh_sw    = (unsigned short*)(ws + (22u << 20));  // 4 MB

  // Swizzle-convert weights: one thread per 32B output chunk.
  const int TW = (kD / 32) * (kD / 16) * 32;            // K=1024
  const int TU = ((kDIN + kD) / 32) * (kD / 16) * 32;   // K=2048
  cvt_swizzle_b_kernel<<<(TW + 255) / 256, 256, 0, stream>>>(w_z, wz_sw, kD);
  cvt_swizzle_b_kernel<<<(TW + 255) / 256, 256, 0, stream>>>(w_r, wr_sw, kD);
  cvt_swizzle_b_kernel<<<(TW + 255) / 256, 256, 0, stream>>>(w_h, wh_sw, kD);
  cvt_swizzle_b_kernel<<<(TU + 255) / 256, 256, 0, stream>>>(u_z, uz_sw, kDIN + kD);
  cvt_swizzle_b_kernel<<<(TU + 255) / 256, 256, 0, stream>>>(u_r, ur_sw, kDIN + kD);
  cvt_swizzle_b_kernel<<<(TU + 255) / 256, 256, 0, stream>>>(u_h, uh_sw, kDIN + kD);

  attn_ctx_kernel<<<kB, 256, 0, stream>>>(x, state, constants, w_att, b_att,
                                          v_f32, state_bf, inp_bf);

  gates_zr_kernel<<<dim3(kD / 128, kB / 16, 2), 256, 0, stream>>>(
      state_bf, inp_bf, wz_sw, uz_sw, wr_sw, ur_sw, b_z, b_r, state,
      z_f32, sr_bf);

  final_h_kernel<<<dim3(kD / 128, kB / 16, 1), 256, 0, stream>>>(
      sr_bf, inp_bf, wh_sw, uh_sw, b_h, z_f32, state, v_f32, out);
}
